// E89ResidualStateCell_45990509805692
// MI455X (gfx1250) — matrix operand order repaired and checked by reading, not compiled, not run on hardware
//
#include <hip/hip_runtime.h>
#include <hip/hip_bf16.h>

// Problem constants (match reference)
#define TLEN   2048
#define BATCH  4
#define DIMX   1024
#define NHEAD  56
#define NSTATE 32
#define KEYDIM 1792              // NHEAD*NSTATE
#define MROWS  (TLEN*BATCH)      // 8192

typedef __attribute__((ext_vector_type(16))) _Float16 v16h;
typedef __attribute__((ext_vector_type(8)))  _Float16 v8h;
typedef __attribute__((ext_vector_type(4)))  _Float16 v4h;
typedef __attribute__((ext_vector_type(8)))  float    v8f;

#if __has_builtin(__builtin_amdgcn_global_load_async_to_lds_b128)
#define HAVE_ASYNC_LDS 1
#endif

// ---------------------------------------------------------------- math helpers
__device__ __forceinline__ float fast_exp2(float x) {
#if __has_builtin(__builtin_amdgcn_exp2f)
  return __builtin_amdgcn_exp2f(x);
#else
  return exp2f(x);
#endif
}
__device__ __forceinline__ float fast_log2(float x) {
#if __has_builtin(__builtin_amdgcn_logf)
  return __builtin_amdgcn_logf(x);
#else
  return log2f(x);
#endif
}
__device__ __forceinline__ float fast_rcp(float x) {
#if __has_builtin(__builtin_amdgcn_rcpf)
  return __builtin_amdgcn_rcpf(x);
#else
  return 1.0f / x;
#endif
}
__device__ __forceinline__ float fast_rsq(float x) {
#if __has_builtin(__builtin_amdgcn_rsqf)
  return __builtin_amdgcn_rsqf(x);
#else
  return rsqrtf(x);
#endif
}
#define LOG2E 1.4426950408889634f
__device__ __forceinline__ float silu_f(float x) {
  return x * fast_rcp(1.0f + fast_exp2(-LOG2E * x));   // 1 exp + 1 rcp
}
__device__ __forceinline__ float tanh_f(float x) {
#if __has_builtin(__builtin_amdgcn_tanhf)
  return __builtin_amdgcn_tanhf(x);          // native V_TANH_F32 on gfx1250
#else
  float cx = fminf(fmaxf(x, -9.0f), 9.0f);
  float e  = fast_exp2(2.0f * LOG2E * cx);   // exp(2x)
  return (e - 1.0f) * fast_rcp(e + 1.0f);
#endif
}
__device__ __forceinline__ float bcast_lane(float x, int i) {
#if __has_builtin(__builtin_amdgcn_readlane)
  return __int_as_float(__builtin_amdgcn_readlane(__float_as_int(x), i));
#else
  return __shfl(x, i, 32);
#endif
}

#ifdef HAVE_ASYNC_LDS
// Builtin signature (from clang): (global v4i*, lds v4i*, imm offset, imm cpol)
typedef __attribute__((__vector_size__(4 * sizeof(int)))) int v4i_t;
typedef __attribute__((address_space(1))) v4i_t* gbl_v4i_p;
typedef __attribute__((address_space(3))) v4i_t* lds_v4i_p;

__device__ __forceinline__ void async_cp16(const _Float16* g, _Float16* l) {
  __builtin_amdgcn_global_load_async_to_lds_b128(
      (gbl_v4i_p)(size_t)g, (lds_v4i_p)(size_t)l, 0, 0);
}
__device__ __forceinline__ void wait_async_le2() {
#if __has_builtin(__builtin_amdgcn_s_wait_asynccnt)
  __builtin_amdgcn_s_wait_asynccnt(2);
#else
  asm volatile("s_wait_asynccnt 0x2" ::: "memory");
#endif
}
__device__ __forceinline__ void wait_async_le0() {
#if __has_builtin(__builtin_amdgcn_s_wait_asynccnt)
  __builtin_amdgcn_s_wait_asynccnt(0);
#else
  asm volatile("s_wait_asynccnt 0x0" ::: "memory");
#endif
}
#endif

// ---------------------------------------------------------------- f32 -> f16 pass
__global__ __launch_bounds__(256) void cvt_f16_kernel(
    const float* __restrict__ src, _Float16* __restrict__ dst, int n4) {
  int i = blockIdx.x * blockDim.x + threadIdx.x;
  if (i >= n4) return;
  float4 u = ((const float4*)src)[i];
  v4h h; h[0]=(_Float16)u.x; h[1]=(_Float16)u.y; h[2]=(_Float16)u.z; h[3]=(_Float16)u.w;
  ((v4h*)dst)[i] = h;
}

// ---------------------------------------------------------------- WMMA GEMM
// Y[m, n0+n] = act( X[m,:] . W[n0+n,:] ),  X:[M,KTOT] f16 row-major,
// W:[N,KTOT] f16 row-major (pre-converted). Block = 4 waves; per wave a
// 32(M) x 64(N) tile, 8x v_wmma_f32_16x16x32_f16 per 32-wide K-step.
// The 64x32 B tile (shared by all 4 waves) is staged in LDS via async
// global->LDS copies, 3-deep pipeline, 1 barrier + 1 asynccnt wait per step
// (peeled prologue/epilogue keep the steady-state loop branch-free).
// ACT: 0 = none, 1 = silu, 2 = silu + l2norm per 32-wide head (2 heads/tile).
template <int KTOT, int ACT>
__global__ __launch_bounds__(128) void gemm32x64_kernel(
    const _Float16* __restrict__ X, const _Float16* __restrict__ W,
    float* __restrict__ out, int ld_out) {
  const int lane = threadIdx.x & 31;
  const int wave = threadIdx.x >> 5;
  const int half = lane >> 4;   // lane-half selects K sub-block
  const int lrow = lane & 15;
  const int m0 = (blockIdx.y * 4 + wave) * 32;
  const int n0 = blockIdx.x * 64;

  const _Float16* xrow[2];
#pragma unroll
  for (int mt = 0; mt < 2; ++mt) xrow[mt] = X + (size_t)(m0 + 16 * mt + lrow) * KTOT;

  v8f acc[2][4];
#pragma unroll
  for (int mt = 0; mt < 2; ++mt)
#pragma unroll
    for (int nt = 0; nt < 4; ++nt) acc[mt][nt] = (v8f){};

#ifdef HAVE_ASYNC_LDS
  // ---------------- async-LDS-staged B path ----------------
  constexpr int NK = KTOT / 32;          // 32 or 56 K-steps
  __shared__ _Float16 lbs[3][64 * 32];   // 3 x 4KB circular staging buffers

  const int tid = threadIdx.x;           // 128 threads stage 256 16B chunks
  // chunk c: row nn = c>>2 (64 rows), 8-f16 sub-block sub = c&3
  auto stage = [&](int ik, int buf) {
#pragma unroll
    for (int c = tid; c < 256; c += 128) {
      const int nn = c >> 2, sub = c & 3;
      async_cp16(W + (size_t)(n0 + nn) * KTOT + ik * 32 + sub * 8,
                 &lbs[buf][nn * 32 + sub * 8]);
    }
  };
  auto compute = [&](int ik, int buf) {
    const int k0 = ik * 32;
    v16h a[2];
#pragma unroll
    for (int mt = 0; mt < 2; ++mt) {
      v8h g0 = *(const v8h*)(xrow[mt] + k0 + 8 * half);
      v8h g1 = *(const v8h*)(xrow[mt] + k0 + 16 + 8 * half);
#pragma unroll
      for (int j = 0; j < 8; ++j) { a[mt][j] = g0[j]; a[mt][8 + j] = g1[j]; }
    }
    const _Float16* bbuf = lbs[buf];
    v16h b[4];
#pragma unroll
    for (int nt = 0; nt < 4; ++nt) {
      v8h g0 = *(const v8h*)(bbuf + (16 * nt + lrow) * 32 + 8 * half);
      v8h g1 = *(const v8h*)(bbuf + (16 * nt + lrow) * 32 + 16 + 8 * half);
#pragma unroll
      for (int j = 0; j < 8; ++j) { b[nt][j] = g0[j]; b[nt][8 + j] = g1[j]; }
    }
#pragma unroll
    for (int mt = 0; mt < 2; ++mt)
#pragma unroll
      for (int nt = 0; nt < 4; ++nt)
        acc[mt][nt] = __builtin_amdgcn_wmma_f32_16x16x32_f16(
            false, a[mt], false, b[nt], (short)0, acc[mt][nt], false, false);
  };

  stage(0, 0);
  stage(1, 1);
  int sbuf = 2, rbuf = 0;
  for (int ik = 0; ik < NK - 2; ++ik) {   // steady state: branch-free body
    wait_async_le2();          // stage ik landed (only ik+1's 2 loads in flight)
    __syncthreads();           // B[ik] visible; all reads of sbuf's buffer done
    stage(ik + 2, sbuf);
    compute(ik, rbuf);
    sbuf = (sbuf == 2) ? 0 : sbuf + 1;
    rbuf = (rbuf == 2) ? 0 : rbuf + 1;
  }
  wait_async_le2();            // stage NK-2 landed (NK-1 may be in flight)
  __syncthreads();
  compute(NK - 2, rbuf);
  rbuf = (rbuf == 2) ? 0 : rbuf + 1;
  wait_async_le0();            // stage NK-1 landed
  __syncthreads();
  compute(NK - 1, rbuf);
#else
  // ---------------- direct-global fallback path ----------------
  const _Float16* wrow[4];
#pragma unroll
  for (int nt = 0; nt < 4; ++nt) wrow[nt] = W + (size_t)(n0 + 16 * nt + lrow) * KTOT;

  for (int k0 = 0; k0 < KTOT; k0 += 32) {
    v16h a[2];
#pragma unroll
    for (int mt = 0; mt < 2; ++mt) {
      v8h g0 = *(const v8h*)(xrow[mt] + k0 + 8 * half);
      v8h g1 = *(const v8h*)(xrow[mt] + k0 + 16 + 8 * half);
#pragma unroll
      for (int j = 0; j < 8; ++j) { a[mt][j] = g0[j]; a[mt][8 + j] = g1[j]; }
    }
    v16h b[4];
#pragma unroll
    for (int nt = 0; nt < 4; ++nt) {
      v8h g0 = *(const v8h*)(wrow[nt] + k0 + 8 * half);
      v8h g1 = *(const v8h*)(wrow[nt] + k0 + 16 + 8 * half);
#pragma unroll
      for (int j = 0; j < 8; ++j) { b[nt][j] = g0[j]; b[nt][8 + j] = g1[j]; }
    }
#pragma unroll
    for (int mt = 0; mt < 2; ++mt)
#pragma unroll
      for (int nt = 0; nt < 4; ++nt)
        acc[mt][nt] = __builtin_amdgcn_wmma_f32_16x16x32_f16(
            false, a[mt], false, b[nt], (short)0, acc[mt][nt], false, false);
  }
#endif

  // epilogue. C layout: VGPR r / lanes 0-15 -> M=r; lanes 16-31 -> M=r+8.
  float c[2][4][8];
#pragma unroll
  for (int mt = 0; mt < 2; ++mt)
#pragma unroll
    for (int nt = 0; nt < 4; ++nt)
#pragma unroll
      for (int r = 0; r < 8; ++r) c[mt][nt][r] = acc[mt][nt][r];

  if constexpr (ACT >= 1) {
#pragma unroll
    for (int mt = 0; mt < 2; ++mt)
#pragma unroll
      for (int nt = 0; nt < 4; ++nt)
#pragma unroll
        for (int r = 0; r < 8; ++r) c[mt][nt][r] = silu_f(c[mt][nt][r]);
  }
  if constexpr (ACT == 2) {
    // l2 norm per head: head0 = nt{0,1}, head1 = nt{2,3}; xor masks <=8 stay
    // inside a 16-lane half (= one matrix row).
#pragma unroll
    for (int mt = 0; mt < 2; ++mt)
#pragma unroll
      for (int hd = 0; hd < 2; ++hd)
#pragma unroll
        for (int r = 0; r < 8; ++r) {
          float s = c[mt][2 * hd][r] * c[mt][2 * hd][r] +
                    c[mt][2 * hd + 1][r] * c[mt][2 * hd + 1][r];
          s += __shfl_xor(s, 1, 32);
          s += __shfl_xor(s, 2, 32);
          s += __shfl_xor(s, 4, 32);
          s += __shfl_xor(s, 8, 32);
          float inv = fast_rsq(fmaxf(s, 1e-24f));   // == 1/max(||x||,1e-12)
          c[mt][2 * hd][r] *= inv;
          c[mt][2 * hd + 1][r] *= inv;
        }
  }
#pragma unroll
  for (int mt = 0; mt < 2; ++mt)
#pragma unroll
    for (int r = 0; r < 8; ++r) {
      int m = m0 + 16 * mt + r + 8 * half;
#pragma unroll
      for (int nt = 0; nt < 4; ++nt)
        out[(size_t)m * ld_out + n0 + 16 * nt + lrow] = c[mt][nt][r];
    }
}

// ---------------------------------------------------------------- decay kernel
// decay[m*H + h] = exp(-exp(A_log[h]) * softplus(x[m,:].Wa[h,:] + dt_bias[h]))
__global__ __launch_bounds__(256) void decay_kernel(
    const float* __restrict__ x, const float* __restrict__ Wa,
    const float* __restrict__ A_log, const float* __restrict__ dt_bias,
    float* __restrict__ decay) {
  int gid = blockIdx.x * (blockDim.x >> 5) + (threadIdx.x >> 5);  // one wave per (m,h)
  if (gid >= MROWS * NHEAD) return;
  int m = gid / NHEAD, h = gid % NHEAD;
  int lane = threadIdx.x & 31;
  const float* xr = x + (size_t)m * DIMX;
  const float* wr = Wa + (size_t)h * DIMX;
  float a0 = 0.0f, a1 = 0.0f;
#pragma unroll 4
  for (int cc = lane; cc < DIMX; cc += 64) {
    a0 = fmaf(xr[cc], wr[cc], a0);
    a1 = fmaf(xr[cc + 32], wr[cc + 32], a1);
  }
  float acc = a0 + a1;
  acc += __shfl_xor(acc, 16, 32);
  acc += __shfl_xor(acc, 8, 32);
  acc += __shfl_xor(acc, 4, 32);
  acc += __shfl_xor(acc, 2, 32);
  acc += __shfl_xor(acc, 1, 32);
  if (lane == 0) {
    float a  = acc + dt_bias[h];
    float dt = (a > 20.0f) ? a
             : 0.6931471805599453f * fast_log2(1.0f + fast_exp2(LOG2E * a));
    decay[gid] = fast_exp2(-LOG2E * fast_exp2(LOG2E * A_log[h]) * dt);
  }
}

// ---------------------------------------------------------------- scan kernel
// One wave32 per (b,h). State S[i][j]: lane j holds column j in 32 VGPRs.
// Per-step loads for t+1 are issued at the top of step t (double buffer) so a
// full step of VALU hides the L2 latency. k_i/q_i broadcasts via v_readlane.
__global__ __launch_bounds__(32) void scan_kernel(
    const float* __restrict__ q, const float* __restrict__ k,
    const float* __restrict__ v, const float* __restrict__ dec,
    const float* __restrict__ S0, _Float16* __restrict__ sqh,
    float* __restrict__ Snew) {
  const int bh = blockIdx.x;              // b*NHEAD + h
  const int b = bh / NHEAD;
  const int h = bh % NHEAD;
  const int lane = threadIdx.x & 31;

  float S[NSTATE];
#pragma unroll
  for (int i = 0; i < NSTATE; ++i)
    S[i] = S0[(size_t)bh * (NSTATE * NSTATE) + i * NSTATE + lane];

  const size_t TS = (size_t)BATCH * NHEAD * NSTATE;   // 7168: t-stride, k/q/v
  const size_t base0 = ((size_t)b * NHEAD + h) * NSTATE;
  const float* kp = k + base0 + lane;
  const float* qp = q + base0 + lane;
  const float* vp = v + base0 + lane;
  const float* dp = dec + (size_t)b * NHEAD + h;      // t-stride BATCH*NHEAD
  _Float16* sp = sqh + base0 + lane;

  float kv = kp[0], qv = qp[0], vv = vp[0], dd = dp[0];

  for (int t = 0; t < TLEN; ++t) {
    // ---- prefetch t+1 (consumed only at loop end -> latency hidden)
    float kv_n = 0.0f, qv_n = 0.0f, vv_n = 0.0f, dd_n = 0.0f;
    if (t + 1 < TLEN) {
      const size_t off = (size_t)(t + 1) * TS;
      kv_n = kp[off]; qv_n = qp[off]; vv_n = vp[off];
      dd_n = dp[(size_t)(t + 1) * (BATCH * NHEAD)];
    }

    // ---- broadcast k row once; retrieved_j = sum_i S[i][j]*k_i (4 chains)
    float ks[NSTATE];
    float r0 = 0.0f, r1 = 0.0f, r2 = 0.0f, r3 = 0.0f;
#pragma unroll
    for (int i = 0; i < NSTATE; i += 4) {
      ks[i]     = bcast_lane(kv, i);
      ks[i + 1] = bcast_lane(kv, i + 1);
      ks[i + 2] = bcast_lane(kv, i + 2);
      ks[i + 3] = bcast_lane(kv, i + 3);
      r0 = fmaf(ks[i],     S[i],     r0);
      r1 = fmaf(ks[i + 1], S[i + 1], r1);
      r2 = fmaf(ks[i + 2], S[i + 2], r2);
      r3 = fmaf(ks[i + 3], S[i + 3], r3);
    }
    const float delta = vv - ((r0 + r1) + (r2 + r3));

    // ---- S += tanh(d*S + k (x) delta);  Sq_j = sum_i S[i][j]*q_i (4 chains)
    float s0 = 0.0f, s1 = 0.0f, s2 = 0.0f, s3 = 0.0f;
#pragma unroll
    for (int i = 0; i < NSTATE; i += 4) {
      float u0 = fmaf(ks[i],     delta, dd * S[i]);
      float u1 = fmaf(ks[i + 1], delta, dd * S[i + 1]);
      float u2 = fmaf(ks[i + 2], delta, dd * S[i + 2]);
      float u3 = fmaf(ks[i + 3], delta, dd * S[i + 3]);
      S[i]     += tanh_f(u0);
      S[i + 1] += tanh_f(u1);
      S[i + 2] += tanh_f(u2);
      S[i + 3] += tanh_f(u3);
      s0 = fmaf(bcast_lane(qv, i),     S[i],     s0);
      s1 = fmaf(bcast_lane(qv, i + 1), S[i + 1], s1);
      s2 = fmaf(bcast_lane(qv, i + 2), S[i + 2], s2);
      s3 = fmaf(bcast_lane(qv, i + 3), S[i + 3], s3);
    }
    sp[(size_t)t * TS] = (_Float16)((s0 + s1) + (s2 + s3));

    kv = kv_n; qv = qv_n; vv = vv_n; dd = dd_n;
  }

#pragma unroll
  for (int i = 0; i < NSTATE; ++i)
    Snew[(size_t)bh * (NSTATE * NSTATE) + i * NSTATE + lane] = S[i];
}

// ---------------------------------------------------------------- launch
extern "C" void kernel_launch(void* const* d_in, const int* in_sizes, int n_in,
                              void* d_out, int out_size, void* d_ws, size_t ws_size,
                              hipStream_t stream) {
  const float* x       = (const float*)d_in[0];
  const float* S0      = (const float*)d_in[1];
  const float* Wq      = (const float*)d_in[2];
  const float* Wk      = (const float*)d_in[3];
  const float* Wv      = (const float*)d_in[4];
  const float* Wa      = (const float*)d_in[5];
  const float* A_log   = (const float*)d_in[6];
  const float* dt_bias = (const float*)d_in[7];
  const float* Wout    = (const float*)d_in[8];

  float* outp = (float*)d_out;                     // [T,B,DIM]
  float* Snew = outp + (size_t)MROWS * DIMX;       // [B,H,NS,NS]

  // workspace carve-up
  float* qb  = (float*)d_ws;                                   // [M,KEY] f32
  float* kb  = qb + (size_t)MROWS * KEYDIM;                    // [M,KEY] f32
  float* vb  = kb + (size_t)MROWS * KEYDIM;                    // [M,KEY] f32
  float* dcb = vb + (size_t)MROWS * KEYDIM;                    // [M,H]   f32
  _Float16* xh   = (_Float16*)(dcb + (size_t)MROWS * NHEAD);   // [M,DIM]
  _Float16* wqh  = xh  + (size_t)MROWS * DIMX;                 // [KEY,DIM]
  _Float16* wkh  = wqh + (size_t)KEYDIM * DIMX;
  _Float16* wvh  = wkh + (size_t)KEYDIM * DIMX;
  _Float16* wouth= wvh + (size_t)KEYDIM * DIMX;                // [DIM,KEY]
  _Float16* sqh  = wouth + (size_t)DIMX * KEYDIM;              // [M,KEY]

  // 1) f32 -> f16 pre-conversions (pure bandwidth)
  {
    int n;
    n = MROWS * DIMX / 4;   cvt_f16_kernel<<<(n + 255) / 256, 256, 0, stream>>>(x,    xh,    n);
    n = KEYDIM * DIMX / 4;  cvt_f16_kernel<<<(n + 255) / 256, 256, 0, stream>>>(Wq,   wqh,   n);
    n = KEYDIM * DIMX / 4;  cvt_f16_kernel<<<(n + 255) / 256, 256, 0, stream>>>(Wk,   wkh,   n);
    n = KEYDIM * DIMX / 4;  cvt_f16_kernel<<<(n + 255) / 256, 256, 0, stream>>>(Wv,   wvh,   n);
    n = DIMX * KEYDIM / 4;  cvt_f16_kernel<<<(n + 255) / 256, 256, 0, stream>>>(Wout, wouth, n);
  }

  // 2) projection GEMMs (WMMA f16, f32 accumulate)
  dim3 blk(128);
  dim3 g_qkv(KEYDIM / 64, MROWS / 128);
  gemm32x64_kernel<DIMX, 2><<<g_qkv, blk, 0, stream>>>(xh, wqh, qb, KEYDIM);
  gemm32x64_kernel<DIMX, 2><<<g_qkv, blk, 0, stream>>>(xh, wkh, kb, KEYDIM);
  gemm32x64_kernel<DIMX, 1><<<g_qkv, blk, 0, stream>>>(xh, wvh, vb, KEYDIM);

  // 3) gate/decay projection
  int nwaves = MROWS * NHEAD;
  decay_kernel<<<(nwaves + 7) / 8, 256, 0, stream>>>(x, Wa, A_log, dt_bias, dcb);

  // 4) sequential residual-state scan (one wave per (b,h))
  scan_kernel<<<BATCH * NHEAD, 32, 0, stream>>>(qb, kb, vb, dcb, S0, sqh, Snew);

  // 5) output GEMM
  dim3 g_out(DIMX / 64, MROWS / 128);
  gemm32x64_kernel<KEYDIM, 0><<<g_out, blk, 0, stream>>>(sqh, wouth, outp, DIMX);
}
